// DPConv_35338990911565
// MI455X (gfx1250) — matrix operand order repaired and checked by reading, not compiled, not run on hardware
//
#include <hip/hip_runtime.h>
#include <hip/hip_bf16.h>

typedef __attribute__((ext_vector_type(16))) _Float16 v16h;
typedef __attribute__((ext_vector_type(8)))  float    v8f;

#define WAVES 8
#define TPB   256

union Frag16 { v16h v; uint4 q[2]; };
union Pack8  { uint4 q; _Float16 h[8]; };

// A fragment, row-major A[M][K], 16x32 f16 tile at (m0,k0).
// Lane layout (ISA 7.12.2): M = m0 + (lane&15); elements 0..7 -> K = k0+kh*8+i,
// elements 8..15 -> K = k0+16+kh*8+i, kh = lane>>4.  Two 16B loads.
__device__ __forceinline__ v16h load_a_rm(const _Float16* A, int lda, int m0, int k0, int lane) {
  Frag16 f;
  const _Float16* p = A + (m0 + (lane & 15)) * lda + k0 + ((lane >> 4) << 3);
  f.q[0] = *(const uint4*)(p);
  f.q[1] = *(const uint4*)(p + 16);
  return f.v;
}

// B fragment from B^T stored row-major BT[N][K] (32x16 f16 tile at (k0,n0)).
// Lane: N = n0 + (lane&15); elements 0..7 -> K = k0+kh*8+i, elements 8..15 -> K = k0+16+kh*8+i.  Two 16B loads.
__device__ __forceinline__ v16h load_bt_rm(const _Float16* BT, int ldb, int n0, int k0, int lane) {
  Frag16 f;
  const _Float16* p = BT + (n0 + (lane & 15)) * ldb + k0 + ((lane >> 4) << 3);
  f.q[0] = *(const uint4*)(p);
  f.q[1] = *(const uint4*)(p + 16);
  return f.v;
}

__device__ __forceinline__ v8f wmma_f16(v16h a, v16h b, v8f c) {
  return __builtin_amdgcn_wmma_f32_16x16x32_f16(false, a, false, b, (short)0, c, false, false);
}

__global__ void convert_w_kernel(const float* __restrict__ qkv_w, const float* __restrict__ proj_w,
                                 _Float16* __restrict__ qkvw16, _Float16* __restrict__ projw16) {
  int i = blockIdx.x * blockDim.x + threadIdx.x;
  if (i < 256 * 128) qkvw16[i] = (_Float16)qkv_w[i];
  if (i < 128 * 128) projw16[i] = (_Float16)proj_w[i];
}

__global__ void init_out_kernel(const float* __restrict__ x, float* __restrict__ out, int n) {
  int i = blockIdx.x * blockDim.x + threadIdx.x;
  if (i < n) out[i] = 0.25f * x[i];
}

// One workgroup per patch. Whole attention pipeline LDS-resident.
template<int KK>
__global__ __launch_bounds__(TPB) void attn_kernel(
    const float* __restrict__ x,
    const _Float16* __restrict__ qkvw,   // [256][128] f16
    const float* __restrict__ qkv_b,     // [256]
    const _Float16* __restrict__ projw,  // [128][128] f16
    const float* __restrict__ proj_b,    // [128]
    const float* __restrict__ pe_w,      // [128][3][3]
    const float* __restrict__ pe_b,      // [128]
    float* __restrict__ out)             // [8][128][64][64]
{
  constexpr int N   = KK * KK;               // pixels per patch: 16 / 64 / 144
  constexpr int NP  = (N + 31) & ~31;        // K-padded pixels:  32 / 64 / 160
  constexpr int NT  = N / 16;                // N tiles
  constexpr int NHW = (64 - KK) / 4 + 1;     // patches per dim
  constexpr float SCALE = 0.17677669529663687f;  // 32^-0.5

  extern __shared__ char smem[];
  _Float16* xpT = (_Float16*)smem;             // [N][128]   (x^T; reused later as oT)
  _Float16* qk  = xpT + N * 128;               // [2][N][64] (cols 0..31 = q^T, 32..63 = k^T)
  _Float16* vb  = qk + 2 * N * 64;             // [2][64][NP] (v, zero-padded cols)
  float*    Sb  = (float*)(vb + 2 * 64 * NP);  // [N][N] raw scores (one head at a time)
  _Float16* aP  = (_Float16*)(Sb + N * N);     // [N][NP] attn probs (== B^T for O GEMM)
  float*    peW = (float*)(aP + N * NP);       // [128*9] depthwise conv weights
  float*    peB = peW + 128 * 9;               // [128]
  _Float16* oT  = xpT;                         // [N][128] (o + pe)^T

  const int tid  = threadIdx.x;
  const int lane = tid & 31;
  const int wave = tid >> 5;
  const int kh8  = (lane >> 4) << 3;           // C/D row offset for upper half-wave

  const int pidx = blockIdx.x;
  const int bIdx = pidx / (NHW * NHW);
  const int prem = pidx % (NHW * NHW);
  const int h0 = (prem / NHW) * 4;
  const int w0 = (prem % NHW) * 4;

  // ---- phase 0: zero padded buffers, stage conv weights + patch (transposed, f16) ----
  for (int i = tid; i < 2 * 64 * NP; i += TPB) vb[i] = (_Float16)0.f;
  for (int i = tid; i < N * NP;      i += TPB) aP[i] = (_Float16)0.f;
  for (int i = tid; i < 128 * 9;     i += TPB) peW[i] = pe_w[i];
  for (int i = tid; i < 128;         i += TPB) peB[i] = pe_b[i];
  {
    constexpr int NV = N / 4;                  // float4 chunks per channel (KK % 4 == 0)
    for (int i = tid; i < 128 * NV; i += TPB) {
      int c = i / NV, pix = (i % NV) * 4;
      int ph = pix / KK, pw = pix % KK;        // 4-chunk never crosses a patch row
      float4 xv = *(const float4*)(x + (((size_t)bIdx * 128 + c) * 64 + (h0 + ph)) * 64 + (w0 + pw));
      xpT[(pix + 0) * 128 + c] = (_Float16)xv.x;
      xpT[(pix + 1) * 128 + c] = (_Float16)xv.y;
      xpT[(pix + 2) * 128 + c] = (_Float16)xv.z;
      xpT[(pix + 3) * 128 + c] = (_Float16)xv.w;
    }
  }
  __syncthreads();

  // ---- phase 1: QKV = qkv_w(256x128) @ xp(128xN), scatter to q^T/k^T/v ----
  for (int t = wave; t < 16 * NT; t += WAVES) {
    int mt = t / NT, nt = t % NT;
    v8f acc = {};
#pragma unroll
    for (int k0 = 0; k0 < 128; k0 += 32)
      acc = wmma_f16(load_a_rm(qkvw, 128, mt * 16, k0, lane),
                     load_bt_rm(xpT, 128, nt * 16, k0, lane), acc);
    int pix  = nt * 16 + (lane & 15);
    int row0 = mt * 16 + kh8;                  // 8 consecutive output channels per lane
    float4 b0 = *(const float4*)(qkv_b + row0);
    float4 b1 = *(const float4*)(qkv_b + row0 + 4);
    float bv[8] = {b0.x, b0.y, b0.z, b0.w, b1.x, b1.y, b1.z, b1.w};
    int hh = mt >> 3;                          // head: uniform per tile
    if ((mt & 7) < 4) {                        // q^T | k^T : contiguous -> one b128 store
      Pack8 pk;
#pragma unroll
      for (int j = 0; j < 8; ++j) pk.h[j] = (_Float16)(acc[j] + bv[j]);
      int cc0 = ((mt & 7) << 4) + kh8;
      *(uint4*)(qk + hh * (N * 64) + pix * 64 + cc0) = pk.q;
    } else {                                   // v : row-major [d][pix] scatter
      int d0 = (((mt & 7) - 4) << 4) + kh8;
#pragma unroll
      for (int j = 0; j < 8; ++j)
        vb[hh * (64 * NP) + (d0 + j) * NP + pix] = (_Float16)(acc[j] + bv[j]);
    }
  }
  __syncthreads();

  // ---- phase 2 (per head): S = q^T k ; softmax ; O = v @ attn^T ----
  for (int hh = 0; hh < 2; ++hh) {
    for (int t = wave; t < NT * NT; t += WAVES) {   // S[i][j], K = 32 (one WMMA)
      int mt = t / NT, nt = t % NT;
      v8f acc = {};
      acc = wmma_f16(load_a_rm(qk + hh * N * 64,      64, mt * 16, 0, lane),
                     load_bt_rm(qk + hh * N * 64 + 32, 64, nt * 16, 0, lane), acc);
      int j0 = nt * 16 + (lane & 15);
#pragma unroll
      for (int j = 0; j < 8; ++j)
        Sb[(mt * 16 + j + kh8) * N + j0] = acc[j];
    }
    __syncthreads();

    for (int i = wave; i < N; i += WAVES) {         // one wave per row
      float m = -3.0e38f;
      for (int jj = lane; jj < N; jj += 32) m = fmaxf(m, Sb[i * N + jj] * SCALE);
#pragma unroll
      for (int off = 16; off; off >>= 1) m = fmaxf(m, __shfl_xor(m, off));
      float s = 0.f;
      for (int jj = lane; jj < N; jj += 32) {
        float e = __expf(Sb[i * N + jj] * SCALE - m);
        aP[i * NP + jj] = (_Float16)e;
        s += e;
      }
#pragma unroll
      for (int off = 16; off; off >>= 1) s += __shfl_xor(s, off);
      float inv = 1.0f / s;
      for (int jj = lane; jj < N; jj += 32)
        aP[i * NP + jj] = (_Float16)((float)aP[i * NP + jj] * inv);
    }
    __syncthreads();

    for (int t = wave; t < 4 * NT; t += WAVES) {    // O[d][i] = sum_m v[d][m] attn[i][m]
      int mt = t / NT, nt = t % NT;
      v8f acc = {};
#pragma unroll
      for (int k0 = 0; k0 < NP; k0 += 32)
        acc = wmma_f16(load_a_rm(vb + hh * 64 * NP, NP, mt * 16, k0, lane),
                       load_bt_rm(aP, NP, nt * 16, k0, lane), acc);
      int pix = nt * 16 + (lane & 15);
      Pack8 pk;
#pragma unroll
      for (int j = 0; j < 8; ++j) pk.h[j] = (_Float16)acc[j];
      *(uint4*)(oT + pix * 128 + hh * 64 + mt * 16 + kh8) = pk.q;  // contiguous b128
    }
    __syncthreads();
  }

  // ---- phase 3: depthwise 3x3 PE conv on v image, += into o^T ----
  for (int i = tid; i < 128 * N; i += TPB) {
    int c = i / N, pix = i % N;
    int ph = pix / KK, pw = pix % KK;
    float s = peB[c];
    const _Float16* vch = vb + (c >> 6) * (64 * NP) + (c & 63) * NP;
#pragma unroll
    for (int dy = -1; dy <= 1; ++dy)
#pragma unroll
      for (int dx = -1; dx <= 1; ++dx) {
        int yy = ph + dy, xx = pw + dx;
        if (yy >= 0 && yy < KK && xx >= 0 && xx < KK)
          s += (float)vch[yy * KK + xx] * peW[c * 9 + (dy + 1) * 3 + (dx + 1)];
      }
    oT[pix * 128 + c] = (_Float16)((float)oT[pix * 128 + c] + s);
  }
  __syncthreads();

  // ---- phase 4: y = proj_w @ (o+pe), fold with 0.25/count weighting ----
  for (int t = wave; t < 8 * NT; t += WAVES) {
    int mt = t / NT, nt = t % NT;
    v8f acc = {};
#pragma unroll
    for (int k0 = 0; k0 < 128; k0 += 32)
      acc = wmma_f16(load_a_rm(projw, 128, mt * 16, k0, lane),
                     load_bt_rm(oT, 128, nt * 16, k0, lane), acc);
    int pix = nt * 16 + (lane & 15);
    int ph = pix / KK, pw = pix % KK;
    int gh = h0 + ph, gw = w0 + pw;
    int ch = 0, cw = 0;
    for (int s2 = 0; s2 < NHW; ++s2) {        // overlap counts (separable)
      ch += (gh >= s2 * 4 && gh < s2 * 4 + KK);
      cw += (gw >= s2 * 4 && gw < s2 * 4 + KK);
    }
    float cf = 0.25f / (float)(ch * cw);
    int c0 = mt * 16 + kh8;                   // 8 consecutive output channels per lane
    float4 p0 = *(const float4*)(proj_b + c0);
    float4 p1 = *(const float4*)(proj_b + c0 + 4);
    float pb[8] = {p0.x, p0.y, p0.z, p0.w, p1.x, p1.y, p1.z, p1.w};
#pragma unroll
    for (int j = 0; j < 8; ++j) {
      float val = (acc[j] + pb[j]) * cf;
      unsafeAtomicAdd(out + (((size_t)bIdx * 128 + (c0 + j)) * 64 + gh) * 64 + gw, val);
    }
  }
}

extern "C" void kernel_launch(void* const* d_in, const int* in_sizes, int n_in,
                              void* d_out, int out_size, void* d_ws, size_t ws_size,
                              hipStream_t stream) {
  const float* x      = (const float*)d_in[0];
  const float* qkv_b  = (const float*)d_in[2];
  const float* proj_b = (const float*)d_in[4];
  const float* pe_w   = (const float*)d_in[5];
  const float* pe_b   = (const float*)d_in[6];
  float* out = (float*)d_out;

  _Float16* qkvw16  = (_Float16*)d_ws;
  _Float16* projw16 = qkvw16 + 256 * 128;

  convert_w_kernel<<<(256 * 128 + 255) / 256, 256, 0, stream>>>(
      (const float*)d_in[1], (const float*)d_in[3], qkvw16, projw16);

  int total = 8 * 128 * 64 * 64;
  init_out_kernel<<<(total + 255) / 256, 256, 0, stream>>>(x, out, total);

  auto smem = [](int KK) -> size_t {
    size_t n = (size_t)KK * KK;
    size_t NP = (n + 31) & ~(size_t)31;
    return n * 128 * 2        // xpT / oT
         + 2 * n * 64 * 2     // q^T | k^T
         + 2 * 64 * NP * 2    // v (padded)
         + n * n * 4          // S (f32)
         + n * NP * 2         // attn probs (padded)
         + (128 * 9 + 128) * 4; // pe weights/bias
  };

  attn_kernel<4> <<<8 * 16 * 16, TPB, smem(4),  stream>>>(x, qkvw16, qkv_b, projw16, proj_b, pe_w, pe_b, out);
  attn_kernel<8> <<<8 * 15 * 15, TPB, smem(8),  stream>>>(x, qkvw16, qkv_b, projw16, proj_b, pe_w, pe_b, out);
  attn_kernel<12><<<8 * 14 * 14, TPB, smem(12), stream>>>(x, qkvw16, qkv_b, projw16, proj_b, pe_w, pe_b, out);
}
